// GAT_Transformer_61512521613719
// MI455X (gfx1250) — matrix operand order repaired and checked by reading, compiled, not run on hardware
//
#include <hip/hip_runtime.h>
#include <cmath>

// ---------------- problem constants (match reference) ----------------
constexpr int NN   = 20000;    // nodes
constexpr int NE   = 320000;   // edges
constexpr int BB   = 2048;     // batch
constexpr int SS   = 48;       // seq
constexpr int SFD  = 32;
constexpr int NFDC = 128;
constexpr int GOC  = 64;
constexpr int GH   = 4;
constexpr int EDC  = 16;
constexpr int DMC  = 128;
constexpr int FFC  = 512;
constexpr int RTOT = BB * SS;  // 98304 rows through the transformer

typedef __attribute__((ext_vector_type(16))) _Float16 v16h;
typedef __attribute__((ext_vector_type(8)))  _Float16 v8h;
typedef __attribute__((ext_vector_type(4)))  _Float16 v4h;
typedef __attribute__((ext_vector_type(8)))  float    v8f;

// ---------------- generic WMMA GEMM: C = act(A_f32 @ B + bias) ----------------
// A:  [M,K] f32 row-major (K must be a multiple of 32; M may be ragged)
// Bt: [N,K] f16 (pre-transposed weights; N must be a multiple of 64)
// C:  [M,N] f32.  Block = 128 threads (4 waves), tile 64x64, K-chunk 32.
__global__ void __launch_bounds__(128)
k_gemm(const float* __restrict__ A, const _Float16* __restrict__ Bt,
       const float* __restrict__ bias, float* __restrict__ C,
       int M, int N, int K, int act)
{
    __shared__ __align__(32) _Float16 sA[64 * 32];   // [row][k]
    __shared__ __align__(32) _Float16 sBt[64 * 32];  // [n][k]

    const int t    = threadIdx.x;
    const int wave = t >> 5;
    const int lane = t & 31;
    const int l16  = lane & 15;
    const int hi   = lane >> 4;        // 0: lanes 0-15, 1: lanes 16-31
    const int row0 = blockIdx.y * 64;
    const int col0 = blockIdx.x * 64;

    v8f acc[4];
    for (int i = 0; i < 4; ++i)
        for (int j = 0; j < 8; ++j) acc[i][j] = 0.0f;

    for (int k0 = 0; k0 < K; k0 += 32) {
        // speculative prefetch of the next A K-slice
        if (k0 + 32 < K) {
            int pr = row0 + (t >> 1);
            if (pr >= M) pr = M - 1;
            __builtin_prefetch(A + (size_t)pr * K + k0 + 32, 0, 1);
        }

        // stage A tile 64x32 (f32 -> f16): 4x float4 per thread, branch-free
        #pragma unroll
        for (int j = 0; j < 4; ++j) {
            int e  = j * 512 + t * 4;       // element index in tile
            int r  = e >> 5, kk = e & 31;
            int gr = row0 + r;
            if (gr >= M) gr = M - 1;        // clamp instead of branch
            const float4 v = *(const float4*)(A + (size_t)gr * K + k0 + kk);
            v4h hv;
            hv[0] = (_Float16)v.x; hv[1] = (_Float16)v.y;
            hv[2] = (_Float16)v.z; hv[3] = (_Float16)v.w;
            *(v4h*)(&sA[e]) = hv;
        }
        // stage B tile: contiguous copy from pre-transposed Bt (v8h = 16B chunks)
        #pragma unroll
        for (int j = 0; j < 2; ++j) {
            int e  = j * 1024 + t * 8;      // half index in tile
            int nn = e >> 5, kk = e & 31;
            const v8h v = *(const v8h*)(Bt + (size_t)(col0 + nn) * K + k0 + kk);
            *(v8h*)(&sBt[e]) = v;
        }
        __syncthreads();

        // A fragment (16-bit A 16x32 lane layout)
        const _Float16* ap = &sA[(wave * 16 + l16) * 32 + hi * 8];
        v8h alo = *(const v8h*)ap;          // k = hi*8 + 0..7
        v8h ahi = *(const v8h*)(ap + 16);   // k = 16 + hi*8 + 0..7
        v16h af;
        #pragma unroll
        for (int i = 0; i < 8; ++i) { af[i] = alo[i]; af[i + 8] = ahi[i]; }

        // B fragments: same k order as A (k = hi*8 + 0..7, 16 + hi*8 + 0..7) of column n
        #pragma unroll
        for (int nt = 0; nt < 4; ++nt) {
            const _Float16* bp = &sBt[(nt * 16 + l16) * 32 + hi * 8];
            v8h blo = *(const v8h*)bp;
            v8h bhi = *(const v8h*)(bp + 16);
            v16h bf;
            #pragma unroll
            for (int i = 0; i < 8; ++i) { bf[i] = blo[i]; bf[i + 8] = bhi[i]; }
            acc[nt] = __builtin_amdgcn_wmma_f32_16x16x32_f16(
                false, af, false, bf, (short)0, acc[nt], false, false);
        }
        __syncthreads();
    }

    // epilogue: VGPR r -> M = r (lanes 0-15) / r+8 (lanes 16-31); N = l16
    const int rbase = row0 + wave * 16;
    #pragma unroll
    for (int nt = 0; nt < 4; ++nt) {
        const int col = col0 + nt * 16 + l16;       // always < N (N % 64 == 0)
        const float bv = bias ? bias[col] : 0.0f;
        #pragma unroll
        for (int r = 0; r < 8; ++r) {
            int row = rbase + r + hi * 8;
            if (row < M) {
                float v = acc[nt][r] + bv;
                if (act == 1) v = fmaxf(v, 0.0f);
                C[(size_t)row * N + col] = v;
            }
        }
    }
}

// ---------------- small helper kernels ----------------
__global__ void k_fill(float* p, float v, int n) {
    int i = blockIdx.x * blockDim.x + threadIdx.x;
    if (i < n) p[i] = v;
}

// transpose-convert weights: s [K,N] f32 -> d [N,Kp] f16, zero-padded to Kp
__global__ void k_cvt16t(const float* __restrict__ s, _Float16* __restrict__ d,
                         int K, int N, int Kp) {
    int i = blockIdx.x * blockDim.x + threadIdx.x;
    if (i >= N * Kp) return;
    int n = i / Kp, k = i % Kp;
    d[i] = (k < K) ? (_Float16)s[(size_t)k * N + n] : (_Float16)0.0f;
}

// pad edge_attr [E,16] -> [E,32] (zero high K) so the GEMM has K%32==0
__global__ void k_pad32(const float* __restrict__ s, float* __restrict__ d, int tot) {
    int i = blockIdx.x * blockDim.x + threadIdx.x;
    if (i >= tot) return;
    int e = i >> 5, c = i & 31;
    d[i] = (c < EDC) ? s[(size_t)e * EDC + c] : 0.0f;
}

__device__ inline void atomicMaxF(float* addr, float v) {
    if (v >= 0.0f) atomicMax((int*)addr, __float_as_int(v));
    else           atomicMin((unsigned int*)addr, __float_as_uint(v));
}

// fold lin_edge_w [64,256] with att_edge [4,64] -> Wae [64,4]
__global__ void k_fold_edge(const float* __restrict__ lew,
                            const float* __restrict__ ae, float* Wae) {
    int t = threadIdx.x;           // 256
    int c = t >> 2, h = t & 3;
    float s = 0.0f;
    for (int o = 0; o < GOC; ++o) s += lew[c * 256 + h * 64 + o] * ae[h * 64 + o];
    Wae[c * 4 + h] = s;
}

// a_src/a_dst[n,h] = sum_c xw[n,h,c] * att[h,c]
__global__ void k_ad(const float* __restrict__ xw, const float* __restrict__ as,
                     const float* __restrict__ ad, float* a_src, float* a_dst, int n4) {
    int i = blockIdx.x * blockDim.x + threadIdx.x;
    if (i >= n4) return;
    int n = i >> 2, h = i & 3;
    const float* row = xw + (size_t)n * 256 + h * 64;
    float s1 = 0.0f, s2 = 0.0f;
    for (int c = 0; c < 64; ++c) { s1 += row[c] * as[h * 64 + c]; s2 += row[c] * ad[h * 64 + c]; }
    a_src[i] = s1; a_dst[i] = s2;
}

// deg + loop-attr numerator (segment sums over dst)
__global__ void k_degloop(const int* __restrict__ dst, const float* __restrict__ te,
                          float* deg, float* loopsum, int tot) {
    int i = blockIdx.x * blockDim.x + threadIdx.x;
    if (i >= tot) return;
    int e = i >> 6, c = i & 63;
    int d = dst[e];
    atomicAdd(&loopsum[(size_t)d * 64 + c], te[(size_t)e * 64 + c]);
    if (c == 0) atomicAdd(&deg[d], 1.0f);
}

__global__ void k_loopdiv(float* loopat, const float* deg, int tot) {
    int i = blockIdx.x * blockDim.x + threadIdx.x;
    if (i >= tot) return;
    loopat[i] /= fmaxf(deg[i >> 6], 1.0f);
}

// a_edge[(e,h)] = te_or_loop[e] . Wae[:,h]
__global__ void k_aedge(const float* __restrict__ te, const float* __restrict__ loopat,
                        const float* __restrict__ Wae, float* a_edge, int tot4) {
    int i = blockIdx.x * blockDim.x + threadIdx.x;
    if (i >= tot4) return;
    int e = i >> 2, h = i & 3;
    const float* row = (e < NE) ? te + (size_t)e * 64 : loopat + (size_t)(e - NE) * 64;
    float s = 0.0f;
    for (int c = 0; c < 64; ++c) s += row[c] * Wae[c * 4 + h];
    a_edge[i] = s;
}

__global__ void k_alphamax(const int* __restrict__ src, const int* __restrict__ dst,
                           const float* __restrict__ a_src, const float* __restrict__ a_dst,
                           const float* __restrict__ a_edge, float* alpha, float* amax, int tot4) {
    int i = blockIdx.x * blockDim.x + threadIdx.x;
    if (i >= tot4) return;
    int e = i >> 2, h = i & 3;
    int s = (e < NE) ? src[e] : e - NE;
    int d = (e < NE) ? dst[e] : e - NE;
    float al = a_src[s * 4 + h] + a_dst[d * 4 + h] + a_edge[i];
    al = (al > 0.0f) ? al : 0.2f * al;   // leaky relu
    alpha[i] = al;
    atomicMaxF(&amax[d * 4 + h], al);
}

__global__ void k_expden(const int* __restrict__ dst, const float* __restrict__ amax,
                         float* alpha, float* den, int tot4) {
    int i = blockIdx.x * blockDim.x + threadIdx.x;
    if (i >= tot4) return;
    int e = i >> 2, h = i & 3;
    int d = (e < NE) ? dst[e] : e - NE;
    float ex = __expf(alpha[i] - amax[d * 4 + h]);
    alpha[i] = ex;
    atomicAdd(&den[d * 4 + h], ex);
}

__global__ void k_scatter(const int* __restrict__ src, const int* __restrict__ dst,
                          const float* __restrict__ alpha, const float* __restrict__ den,
                          const float* __restrict__ xw, float* gatacc, int tot4) {
    int i = blockIdx.x * blockDim.x + threadIdx.x;
    if (i >= tot4) return;
    int e = i >> 2, h = i & 3;
    int s = (e < NE) ? src[e] : e - NE;
    int d = (e < NE) ? dst[e] : e - NE;
    float w = alpha[i] / den[d * 4 + h];
    const float* xr = xw + (size_t)s * 256 + h * 64;
    float* outp = gatacc + (size_t)d * 256 + h * 64;
    for (int c = 0; c < 64; ++c) atomicAdd(&outp[c], w * xr[c]);
}

// head mean + bias + relu -> gAll[n, off + c]
__global__ void k_gfinal(const float* __restrict__ gatacc, const float* __restrict__ bias,
                         float* gAll, int off, int tot) {
    int i = blockIdx.x * blockDim.x + threadIdx.x;
    if (i >= tot) return;
    int n = i >> 6, c = i & 63;
    const float* g = gatacc + (size_t)n * 256;
    float s = (g[c] + g[64 + c] + g[128 + c] + g[192 + c]) * 0.25f + bias[c];
    gAll[(size_t)n * 128 + off + c] = fmaxf(s, 0.0f);
}

// fusedin[b,s,:] = [gAll[idx[b]] (128) | sequences[b,s] (32)]
__global__ void k_fusedin(const int* __restrict__ idx, const float* __restrict__ gAll,
                          const float* __restrict__ seq, float* fusedin, int tot) {
    int i = blockIdx.x * blockDim.x + threadIdx.x;
    if (i >= tot) return;
    int b = i / (SS * 160);
    int r = i % (SS * 160);
    int s = r / 160, c = r % 160;
    float v = (c < 128) ? gAll[(size_t)idx[b] * 128 + c]
                        : seq[((size_t)b * SS + s) * SFD + (c - 128)];
    fusedin[i] = v;
}

__global__ void k_posenc(float* X, int tot) {
    int i = blockIdx.x * blockDim.x + threadIdx.x;
    if (i >= tot) return;
    int row = i >> 7, d = i & 127;
    int s = row % SS;
    int ii = d >> 1;
    float ang = (float)s * __expf(-(2.0f * (float)ii / 128.0f) * 9.210340372f); // ln(10000)
    X[i] += (d & 1) ? __cosf(ang) : __sinf(ang);
}

// attention: one block per batch element; K/V staged in LDS, q in registers
__global__ void __launch_bounds__(256)
k_attn(const float* __restrict__ Q, const float* __restrict__ Kc,
       const float* __restrict__ V, float* __restrict__ O)
{
    __shared__ float sk[SS * DMC];   // 24 KB
    __shared__ float sv[SS * DMC];   // 24 KB
    const int b = blockIdx.x;
    const int t = threadIdx.x;
    const size_t base = (size_t)b * SS * DMC;
    for (int j = t; j < SS * DMC; j += 256) { sk[j] = Kc[base + j]; sv[j] = V[base + j]; }
    __syncthreads();
    if (t < GH * SS) {
        const int h = t / SS, i = t % SS;
        float q[32];
        const float* qp = Q + base + (size_t)i * DMC + h * 32;
        for (int d = 0; d < 32; ++d) q[d] = qp[d];
        const float scale = 0.1767766953f;  // 1/sqrt(32)
        float m = -3.4e38f;
        for (int j = 0; j < SS; ++j) {
            const float* kp = sk + j * DMC + h * 32;
            float s = 0.0f;
            for (int d = 0; d < 32; ++d) s += q[d] * kp[d];
            m = fmaxf(m, s * scale);
        }
        float den = 0.0f, o[32];
        for (int d = 0; d < 32; ++d) o[d] = 0.0f;
        for (int j = 0; j < SS; ++j) {
            const float* kp = sk + j * DMC + h * 32;
            float s = 0.0f;
            for (int d = 0; d < 32; ++d) s += q[d] * kp[d];
            float p = __expf(s * scale - m);
            den += p;
            const float* vp = sv + j * DMC + h * 32;
            for (int d = 0; d < 32; ++d) o[d] += p * vp[d];
        }
        float inv = 1.0f / den;
        float* op = O + base + (size_t)i * DMC + h * 32;
        for (int d = 0; d < 32; ++d) op[d] = o[d] * inv;
    }
}

// y = LN(x + r) * g + b  (D = 128, one block per row)
__global__ void __launch_bounds__(128)
k_addln(const float* __restrict__ x, const float* __restrict__ r,
        const float* __restrict__ g, const float* __restrict__ b, float* y)
{
    __shared__ float red[128];
    const int row = blockIdx.x, t = threadIdx.x;
    float v = x[(size_t)row * 128 + t] + r[(size_t)row * 128 + t];
    red[t] = v; __syncthreads();
    for (int s = 64; s > 0; s >>= 1) { if (t < s) red[t] += red[t + s]; __syncthreads(); }
    float mu = red[0] * (1.0f / 128.0f); __syncthreads();
    float dv = v - mu;
    red[t] = dv * dv; __syncthreads();
    for (int s = 64; s > 0; s >>= 1) { if (t < s) red[t] += red[t + s]; __syncthreads(); }
    float var = red[0] * (1.0f / 128.0f);
    y[(size_t)row * 128 + t] = dv * rsqrtf(var + 1e-5f) * g[t] + b[t];
}

__global__ void k_final(const float* __restrict__ X2, const float* __restrict__ ow,
                        const float* __restrict__ ob, float* out, int R) {
    int r = blockIdx.x * blockDim.x + threadIdx.x;
    if (r >= R) return;
    float s = ob[0];
    const float* row = X2 + (size_t)r * 128;
    for (int d = 0; d < 128; ++d) s += row[d] * ow[d];
    out[r] = s;
}

// ---------------- host orchestration ----------------
static inline size_t align256(size_t x) { return (x + 255) & ~(size_t)255; }

extern "C" void kernel_launch(void* const* d_in, const int* in_sizes, int n_in,
                              void* d_out, int out_size, void* d_ws, size_t ws_size,
                              hipStream_t stream) {
    (void)in_sizes; (void)n_in; (void)out_size; (void)ws_size;

    const float* seq   = (const float*)d_in[0];
    const int*   eidx  = (const int*)d_in[1];
    const float* eattr = (const float*)d_in[2];
    const float* nfeat = (const float*)d_in[3];
    const int*   nidx  = (const int*)d_in[4];
    const float* ew    = (const float*)d_in[5];
    const float* eb    = (const float*)d_in[6];
    const float* g_lin_w[2] = {(const float*)d_in[7],  (const float*)d_in[13]};
    const float* g_lew[2]   = {(const float*)d_in[8],  (const float*)d_in[14]};
    const float* g_asrc[2]  = {(const float*)d_in[9],  (const float*)d_in[15]};
    const float* g_adst[2]  = {(const float*)d_in[10], (const float*)d_in[16]};
    const float* g_aed[2]   = {(const float*)d_in[11], (const float*)d_in[17]};
    const float* g_bias[2]  = {(const float*)d_in[12], (const float*)d_in[18]};
    const float* fw  = (const float*)d_in[19];
    const float* fb  = (const float*)d_in[20];
    const float* wq  = (const float*)d_in[21];
    const float* bq  = (const float*)d_in[22];
    const float* wk  = (const float*)d_in[23];
    const float* bk  = (const float*)d_in[24];
    const float* wv  = (const float*)d_in[25];
    const float* bv  = (const float*)d_in[26];
    const float* wo  = (const float*)d_in[27];
    const float* bo  = (const float*)d_in[28];
    const float* ln1g = (const float*)d_in[29];
    const float* ln1b = (const float*)d_in[30];
    const float* w1   = (const float*)d_in[31];
    const float* b1f  = (const float*)d_in[32];
    const float* w2   = (const float*)d_in[33];
    const float* b2f  = (const float*)d_in[34];
    const float* ln2g = (const float*)d_in[35];
    const float* ln2b = (const float*)d_in[36];
    const float* ow   = (const float*)d_in[37];
    const float* ob   = (const float*)d_in[38];

    const int* srcp = eidx;
    const int* dstp = eidx + NE;

    // bump allocator over d_ws
    char* ws = (char*)d_ws;
    size_t off = 0;
    auto alloc = [&](size_t bytes) -> void* { void* p = ws + off; off = align256(off + bytes); return p; };

    // pre-transposed f16 weights (Bt layout [N, Kpadded])
    _Float16* ew_h   = (_Float16*)alloc((size_t)GOC * 32 * 2);        // [64,32]  (K=16 padded)
    _Float16* linw_h = (_Float16*)alloc((size_t)256 * NFDC * 2);      // [256,128]
    _Float16* fw_h   = (_Float16*)alloc((size_t)DMC * 160 * 2);       // [128,160]
    _Float16* wq_h   = (_Float16*)alloc((size_t)DMC * DMC * 2);
    _Float16* wk_h   = (_Float16*)alloc((size_t)DMC * DMC * 2);
    _Float16* wv_h   = (_Float16*)alloc((size_t)DMC * DMC * 2);
    _Float16* wo_h   = (_Float16*)alloc((size_t)DMC * DMC * 2);
    _Float16* w1_h   = (_Float16*)alloc((size_t)FFC * DMC * 2);       // [512,128]
    _Float16* w2_h   = (_Float16*)alloc((size_t)DMC * FFC * 2);       // [128,512]

    float* eapad   = (float*)alloc((size_t)NE * 32 * 4);              // padded edge_attr
    float* te      = (float*)alloc((size_t)NE * 64 * 4);
    float* Wae     = (float*)alloc(256 * 4);
    float* xw      = (float*)alloc((size_t)NN * 256 * 4);
    float* a_src   = (float*)alloc((size_t)NN * 4 * 4);
    float* a_dst   = (float*)alloc((size_t)NN * 4 * 4);
    float* a_edge  = (float*)alloc((size_t)(NE + NN) * 4 * 4);
    float* alpha   = (float*)alloc((size_t)(NE + NN) * 4 * 4);
    float* deg     = (float*)alloc((size_t)NN * 4);
    float* loopat  = (float*)alloc((size_t)NN * 64 * 4);
    float* amax    = (float*)alloc((size_t)NN * 4 * 4);
    float* den     = (float*)alloc((size_t)NN * 4 * 4);
    float* gatacc  = (float*)alloc((size_t)NN * 256 * 4);
    float* gAll    = (float*)alloc((size_t)NN * 128 * 4);
    float* fusedin = (float*)alloc((size_t)RTOT * 160 * 4);
    float* X       = (float*)alloc((size_t)RTOT * DMC * 4);
    float* Qb      = (float*)alloc((size_t)RTOT * DMC * 4);
    float* Kb      = (float*)alloc((size_t)RTOT * DMC * 4);
    float* Vb      = (float*)alloc((size_t)RTOT * DMC * 4);
    float* Ob      = (float*)alloc((size_t)RTOT * DMC * 4);
    float* Pb      = (float*)alloc((size_t)RTOT * DMC * 4);
    float* X1      = (float*)alloc((size_t)RTOT * DMC * 4);
    float* Fb      = (float*)alloc((size_t)RTOT * FFC * 4);
    float* G2b     = (float*)alloc((size_t)RTOT * DMC * 4);
    float* X2      = (float*)alloc((size_t)RTOT * DMC * 4);

    auto g1 = [](int n) { return dim3((unsigned)((n + 255) / 256)); };
    const dim3 b256(256);

    // ---- edge attr transform: te = edge_attr @ ew + eb (WMMA, K padded 16->32) ----
    k_pad32<<<g1(NE * 32), b256, 0, stream>>>(eattr, eapad, NE * 32);
    k_cvt16t<<<g1(GOC * 32), b256, 0, stream>>>(ew, ew_h, EDC, GOC, 32);
    k_gemm<<<dim3(1, NE / 64), dim3(128), 0, stream>>>(eapad, ew_h, eb, te, NE, GOC, 32, 0);

    // ---- degree + self-loop edge-attr mean (shared by both GAT layers) ----
    k_fill<<<g1(NN), b256, 0, stream>>>(deg, 0.0f, NN);
    k_fill<<<g1(NN * 64), b256, 0, stream>>>(loopat, 0.0f, NN * 64);
    k_degloop<<<g1(NE * 64), b256, 0, stream>>>(dstp, te, deg, loopat, NE * 64);
    k_loopdiv<<<g1(NN * 64), b256, 0, stream>>>(loopat, deg, NN * 64);

    const int tot4 = (NE + NN) * 4;
    for (int L = 0; L < 2; ++L) {
        // xw = node_features @ lin_w (WMMA, [20000,128]@[128,256])
        k_cvt16t<<<g1(256 * NFDC), b256, 0, stream>>>(g_lin_w[L], linw_h, NFDC, 256, NFDC);
        k_gemm<<<dim3(4, (NN + 63) / 64), dim3(128), 0, stream>>>(nfeat, linw_h, nullptr, xw, NN, 256, NFDC, 0);
        // fold lin_edge_w with att_edge -> Wae [64,4] (removes the 22 GF ewf GEMM)
        k_fold_edge<<<dim3(1), b256, 0, stream>>>(g_lew[L], g_aed[L], Wae);
        k_ad<<<g1(NN * 4), b256, 0, stream>>>(xw, g_asrc[L], g_adst[L], a_src, a_dst, NN * 4);
        k_aedge<<<g1(tot4), b256, 0, stream>>>(te, loopat, Wae, a_edge, tot4);
        // edge softmax over incoming edges (incl. self-loops)
        k_fill<<<g1(NN * 4), b256, 0, stream>>>(amax, -INFINITY, NN * 4);
        k_fill<<<g1(NN * 4), b256, 0, stream>>>(den, 0.0f, NN * 4);
        k_fill<<<g1(NN * 256), b256, 0, stream>>>(gatacc, 0.0f, NN * 256);
        k_alphamax<<<g1(tot4), b256, 0, stream>>>(srcp, dstp, a_src, a_dst, a_edge, alpha, amax, tot4);
        k_expden<<<g1(tot4), b256, 0, stream>>>(dstp, amax, alpha, den, tot4);
        k_scatter<<<g1(tot4), b256, 0, stream>>>(srcp, dstp, alpha, den, xw, gatacc, tot4);
        k_gfinal<<<g1(NN * 64), b256, 0, stream>>>(gatacc, g_bias[L], gAll, L * 64, NN * 64);
    }

    // ---- fuse graph + sequence features, project to d_model (WMMA, K=160) ----
    k_fusedin<<<g1(RTOT * 160), b256, 0, stream>>>(nidx, gAll, seq, fusedin, RTOT * 160);
    k_cvt16t<<<g1(DMC * 160), b256, 0, stream>>>(fw, fw_h, 160, DMC, 160);
    k_gemm<<<dim3(2, RTOT / 64), dim3(128), 0, stream>>>(fusedin, fw_h, fb, X, RTOT, DMC, 160, 0);
    k_posenc<<<g1(RTOT * DMC), b256, 0, stream>>>(X, RTOT * DMC);

    // ---- attention block ----
    k_cvt16t<<<g1(DMC * DMC), b256, 0, stream>>>(wq, wq_h, DMC, DMC, DMC);
    k_cvt16t<<<g1(DMC * DMC), b256, 0, stream>>>(wk, wk_h, DMC, DMC, DMC);
    k_cvt16t<<<g1(DMC * DMC), b256, 0, stream>>>(wv, wv_h, DMC, DMC, DMC);
    k_cvt16t<<<g1(DMC * DMC), b256, 0, stream>>>(wo, wo_h, DMC, DMC, DMC);
    k_gemm<<<dim3(2, RTOT / 64), dim3(128), 0, stream>>>(X, wq_h, bq, Qb, RTOT, DMC, DMC, 0);
    k_gemm<<<dim3(2, RTOT / 64), dim3(128), 0, stream>>>(X, wk_h, bk, Kb, RTOT, DMC, DMC, 0);
    k_gemm<<<dim3(2, RTOT / 64), dim3(128), 0, stream>>>(X, wv_h, bv, Vb, RTOT, DMC, DMC, 0);
    k_attn<<<dim3(BB), b256, 0, stream>>>(Qb, Kb, Vb, Ob);
    k_gemm<<<dim3(2, RTOT / 64), dim3(128), 0, stream>>>(Ob, wo_h, bo, Pb, RTOT, DMC, DMC, 0);
    k_addln<<<dim3(RTOT), dim3(128), 0, stream>>>(X, Pb, ln1g, ln1b, X1);

    // ---- feed-forward (WMMA, ReLU fused into FF1 epilogue) ----
    k_cvt16t<<<g1(FFC * DMC), b256, 0, stream>>>(w1, w1_h, DMC, FFC, DMC);
    k_cvt16t<<<g1(DMC * FFC), b256, 0, stream>>>(w2, w2_h, FFC, DMC, FFC);
    k_gemm<<<dim3(8, RTOT / 64), dim3(128), 0, stream>>>(X1, w1_h, b1f, Fb, RTOT, FFC, DMC, 1);
    k_gemm<<<dim3(2, RTOT / 64), dim3(128), 0, stream>>>(Fb, w2_h, b2f, G2b, RTOT, DMC, FFC, 0);
    k_addln<<<dim3(RTOT), dim3(128), 0, stream>>>(X1, G2b, ln2g, ln2b, X2);

    // ---- output head ----
    k_final<<<g1(RTOT), b256, 0, stream>>>(X2, ow, ob, (float*)d_out, RTOT);
}
